// GNN_QNetwork_29515015258641
// MI455X (gfx1250) — matrix operand order repaired and checked by reading, not compiled, not run on hardware
//
#include <hip/hip_runtime.h>
#include <hip/hip_bf16.h>

typedef __attribute__((ext_vector_type(16))) _Float16 v16h;
typedef __attribute__((ext_vector_type(8)))  float    v8f;

#define NN  50000
#define NE  800000
#define NE2 850000   // NE + NN self loops
#define CC  256
#define NG  8

// ---------------------------------------------------------------------------
// Generic WMMA GEMM: C[M,NT] = act( A[M,K] @ B[K,NT] + bias[n] + rowAdd[rowIdx[m], n] )
// fp32 in/out, f16 WMMA compute (v_wmma_f32_16x16x32_f16), f32 accumulate.
// NT (column count) is a compile-time constant so all B/C row strides fold
// into immediate load/store offsets (no per-element 64-bit address math).
// Block = 256 threads = 8 waves arranged 4(M) x 2(N); each wave owns a
// 16(M) x 32(N) tile (two accumulators sharing one A fragment per k-step),
// so the block tile is 64 x 64 and each k-iteration issues 2 WMMAs.
// K must be a multiple of 32. M,NT guarded (loads clamped so EXEC stays full
// for WMMA; stores bounds-checked). Epilogue variants are compile-time.
// ---------------------------------------------------------------------------
template <int NT, bool HAS_BIAS, bool HAS_ROWADD, bool RELU>
__global__ void gemm_wmma_kernel(const float* __restrict__ A,
                                 const float* __restrict__ B,
                                 float* __restrict__ C,
                                 int M, int K,
                                 const float* __restrict__ bias,
                                 const float* __restrict__ rowAdd,
                                 const int* __restrict__ rowIdx)
{
    const int lane = threadIdx.x & 31;
    const int wave = threadIdx.x >> 5;
    const int wm   = wave >> 1;            // 0..3
    const int wn   = wave & 1;             // 0..1
    const int m0   = blockIdx.y * 64 + wm * 16;
    const int n0   = blockIdx.x * 64 + wn * 32;
    const int grp  = lane >> 4;            // half-wave group
    const int l15  = lane & 15;

    int mload  = m0 + l15;      if (mload  >= M)  mload  = M - 1;   // clamp: keep EXEC full
    int nload0 = n0 + l15;      if (nload0 >= NT) nload0 = NT - 1;
    int nload1 = n0 + 16 + l15; if (nload1 >= NT) nload1 = NT - 1;

    const float* arow = A + (size_t)mload * K;
    // per-lane B base pointers; row offsets i*NT are compile-time immediates
    const float* bp0 = B + (size_t)(grp * 16) * NT + nload0;
    const float* bp1 = B + (size_t)(grp * 16) * NT + nload1;

    v8f acc0 = {}, acc1 = {};
    #pragma unroll 2
    for (int k = 0; k < K; k += 32) {
        __builtin_prefetch(arow + k + 64, 0, 1);   // global_prefetch_b8
        v16h a, b0, b1;
        // A fragment 16x32 f16: lane<16 -> K 0..15 ; lane>=16 -> K 16..31 (matches B)
        #pragma unroll
        for (int i = 0; i < 8; ++i) {
            a[i]     = (_Float16)arow[k + grp * 16 + i];
            a[i + 8] = (_Float16)arow[k + grp * 16 + 8 + i];
        }
        // B fragment 32x16: lane group g holds K = g*16 + i, column = lane&15
        #pragma unroll
        for (int i = 0; i < 16; ++i) {
            b0[i] = (_Float16)bp0[i * NT];
            b1[i] = (_Float16)bp1[i * NT];
        }
        bp0 += 32 * NT;
        bp1 += 32 * NT;
        acc0 = __builtin_amdgcn_wmma_f32_16x16x32_f16(false, a, false, b0,
                                                      (short)0, acc0, false, false);
        acc1 = __builtin_amdgcn_wmma_f32_16x16x32_f16(false, a, false, b1,
                                                      (short)0, acc1, false, false);
    }

    // C/D layout: VGPR v -> row m0 + v + 8*grp, col (sub-tile n base) + (lane&15)
    #pragma unroll
    for (int t = 0; t < 2; ++t) {
        const int nn = n0 + t * 16 + l15;
        const int nc = nn < NT ? nn : NT - 1;
        #pragma unroll
        for (int v = 0; v < 8; ++v) {
            int mm = m0 + grp * 8 + v;
            int mc = mm < M ? mm : M - 1;
            float r = t == 0 ? acc0[v] : acc1[v];
            if (HAS_BIAS)   r += bias[nc];
            if (HAS_ROWADD) r += rowAdd[(size_t)rowIdx[mc] * NT + nc];
            if (RELU)       r = fmaxf(r, 0.0f);
            if (mm < M && nn < NT) C[(size_t)mm * NT + nn] = r;
        }
    }
}

// ---------------------------------------------------------------------------
// Per-node attention logits: as[n,h] = <h[n,h,:], a_src[h,:]>, same for ad.
// ---------------------------------------------------------------------------
__global__ void alphas_kernel(const float* __restrict__ hlin,
                              const float* __restrict__ a_src,
                              const float* __restrict__ a_dst,
                              float* __restrict__ as_, float* __restrict__ ad_,
                              int n_nodes, int H, int D)
{
    int idx = blockIdx.x * blockDim.x + threadIdx.x;
    if (idx >= n_nodes * H) return;
    int h = idx % H, n = idx / H;
    const float* hp = hlin + (size_t)n * (H * D) + (size_t)h * D;
    float s = 0.f, d = 0.f;
    for (int i = 0; i < D; ++i) {
        float v = hp[i];
        s += v * a_src[h * D + i];
        d += v * a_dst[h * D + i];
    }
    as_[idx] = s; ad_[idx] = d;
}

// monotonic uint encoding of float for atomicMax-based segment max
__device__ __forceinline__ unsigned f2key(float f) {
    unsigned u = __float_as_uint(f);
    return (u & 0x80000000u) ? ~u : (u | 0x80000000u);
}
__device__ __forceinline__ float key2f(unsigned k) {
    return __uint_as_float((k & 0x80000000u) ? (k & 0x7FFFFFFFu) : ~k);
}

__device__ __forceinline__ void edge_endpoints(const int* ei, int e, int& s, int& d) {
    if (e < NE) { s = ei[e]; d = ei[NE + e]; }
    else        { s = d = e - NE; }            // self loop
}

__global__ void edge_max_kernel(const int* __restrict__ ei,
                                const float* __restrict__ as_,
                                const float* __restrict__ ad_,
                                unsigned* __restrict__ menc, int H)
{
    int idx = blockIdx.x * blockDim.x + threadIdx.x;
    if (idx >= NE2 * H) return;
    int e = idx / H, h = idx - e * H;
    int s, d; edge_endpoints(ei, e, s, d);
    float v = as_[s * H + h] + ad_[d * H + h];
    v = v > 0.f ? v : 0.2f * v;                // leaky_relu(0.2)
    atomicMax(&menc[d * H + h], f2key(v));
}

__global__ void edge_exp_kernel(const int* __restrict__ ei,
                                const float* __restrict__ as_,
                                const float* __restrict__ ad_,
                                const unsigned* __restrict__ menc,
                                float* __restrict__ denom,
                                float* __restrict__ exbuf, int H)
{
    int idx = blockIdx.x * blockDim.x + threadIdx.x;
    if (idx >= NE2 * H) return;
    int e = idx / H, h = idx - e * H;
    int s, d; edge_endpoints(ei, e, s, d);
    float v = as_[s * H + h] + ad_[d * H + h];
    v = v > 0.f ? v : 0.2f * v;
    float ex = __expf(v - key2f(menc[d * H + h]));
    exbuf[idx] = ex;
    atomicAdd(&denom[d * H + h], ex);
}

// one wave per edge; scatters the 256-wide weighted message with atomics
__global__ void edge_agg_kernel(const int* __restrict__ ei,
                                const float* __restrict__ hlin,
                                const float* __restrict__ exbuf,
                                const float* __restrict__ denom,
                                float* __restrict__ agg, int H, int D)
{
    int gt   = blockIdx.x * blockDim.x + threadIdx.x;
    int e    = gt >> 5;
    int lane = gt & 31;
    if (e >= NE2) return;
    int s, d; edge_endpoints(ei, e, s, d);
    int C = H * D;
    for (int c = lane; c < C; c += 32) {
        int h = c / D;
        float alpha = exbuf[(size_t)e * H + h] / (denom[d * H + h] + 1e-16f);
        atomicAdd(&agg[(size_t)d * C + c], hlin[(size_t)s * C + c] * alpha);
    }
}

// one block per channel: mean/var over nodes -> scale/shift (bias b cancels in BN)
__global__ void bn_stats_kernel(const float* __restrict__ h,
                                const float* __restrict__ gamma,
                                const float* __restrict__ beta,
                                float* __restrict__ scale,
                                float* __restrict__ shift,
                                int n_nodes, int C)
{
    __shared__ float s1[256], s2[256];
    int c = blockIdx.x;
    float a = 0.f, b = 0.f;
    for (int n = threadIdx.x; n < n_nodes; n += blockDim.x) {
        float v = h[(size_t)n * C + c];
        a += v; b += v * v;
    }
    s1[threadIdx.x] = a; s2[threadIdx.x] = b;
    __syncthreads();
    for (int off = blockDim.x >> 1; off > 0; off >>= 1) {
        if ((int)threadIdx.x < off) {
            s1[threadIdx.x] += s1[threadIdx.x + off];
            s2[threadIdx.x] += s2[threadIdx.x + off];
        }
        __syncthreads();
    }
    if (threadIdx.x == 0) {
        float mu  = s1[0] / (float)n_nodes;
        float var = s2[0] / (float)n_nodes - mu * mu;
        float sc  = gamma[c] * __frsqrt_rn(var + 1e-5f);
        scale[c] = sc;
        shift[c] = beta[c] - mu * sc;
    }
}

__global__ void bn_apply_kernel(float* __restrict__ h,
                                const float* __restrict__ scale,
                                const float* __restrict__ shift,
                                int total, int C)
{
    int i = blockIdx.x * blockDim.x + threadIdx.x;
    if (i >= total) return;
    int c = i % C;
    h[i] = fmaxf(h[i] * scale[c] + shift[c], 0.0f);   // BN + ReLU
}

// per-graph sums via block-local LDS accumulation (blockDim must == CC)
__global__ void pool_kernel(const float* __restrict__ h,
                            const int* __restrict__ batch,
                            float* __restrict__ sums,
                            float* __restrict__ cnt, int n_nodes)
{
    __shared__ float acc[NG * CC];
    __shared__ float cacc[NG];
    for (int i = threadIdx.x; i < NG * CC; i += blockDim.x) acc[i] = 0.f;
    if (threadIdx.x < NG) cacc[threadIdx.x] = 0.f;
    __syncthreads();
    int r0 = blockIdx.x * 64;
    int r1 = min(r0 + 64, n_nodes);
    int c  = threadIdx.x;
    for (int r = r0; r < r1; ++r) {
        int b = batch[r];
        acc[b * CC + c] += h[(size_t)r * CC + c];
    }
    if (threadIdx.x == 0)
        for (int r = r0; r < r1; ++r) cacc[batch[r]] += 1.f;
    __syncthreads();
    for (int i = threadIdx.x; i < NG * CC; i += blockDim.x)
        atomicAdd(&sums[i], acc[i]);
    if (threadIdx.x < NG) atomicAdd(&cnt[threadIdx.x], cacc[threadIdx.x]);
}

__global__ void gmean_kernel(const float* __restrict__ sums,
                             const float* __restrict__ cnt,
                             float* __restrict__ gmean)
{
    int i = blockIdx.x * blockDim.x + threadIdx.x;
    if (i >= NG * CC) return;
    gmean[i] = sums[i] / fmaxf(cnt[i / CC], 1.0f);
}

__global__ void tail_kernel(const int* __restrict__ batch,
                            float* __restrict__ out, int n)
{
    int i = blockIdx.x * blockDim.x + threadIdx.x;
    if (i < n) out[i] = (float)batch[i];
}

// ---------------------------------------------------------------------------
extern "C" void kernel_launch(void* const* d_in, const int* in_sizes, int n_in,
                              void* d_out, int out_size, void* d_ws, size_t ws_size,
                              hipStream_t stream)
{
    const float* x      = (const float*)d_in[0];
    const int*   ei     = (const int*)  d_in[1];
    const int*   batch  = (const int*)  d_in[2];
    const float* W1     = (const float*)d_in[3];
    const float* a_src1 = (const float*)d_in[4];
    const float* a_dst1 = (const float*)d_in[5];
    // d_in[6] = b1: cancelled by training-mode BatchNorm (constant per-channel shift)
    const float* g1     = (const float*)d_in[7];
    const float* beta1  = (const float*)d_in[8];
    const float* W2     = (const float*)d_in[9];
    const float* a_src2 = (const float*)d_in[10];
    const float* a_dst2 = (const float*)d_in[11];
    // d_in[12] = b2: cancelled by BN
    const float* g2     = (const float*)d_in[13];
    const float* beta2  = (const float*)d_in[14];
    const float* Wq1    = (const float*)d_in[15];
    const float* bq1    = (const float*)d_in[16];
    const float* Wq2    = (const float*)d_in[17];
    const float* bq2    = (const float*)d_in[18];
    float* qout = (float*)d_out;

    // workspace carve (~120 MB)
    char* p = (char*)d_ws;
    auto carve = [&](size_t elems) -> void* {
        void* r = (void*)p;
        p += ((elems * 4) + 255) & ~(size_t)255;
        return r;
    };
    float*    hlin  = (float*)   carve((size_t)NN * CC);  // h@W of current layer / q1
    float*    agg   = (float*)   carve((size_t)NN * CC);  // aggregated msgs -> BN/ReLU in place
    float*    asv   = (float*)   carve((size_t)NN * 4);
    float*    adv   = (float*)   carve((size_t)NN * 4);
    unsigned* menc  = (unsigned*)carve((size_t)NN * 4);
    float*    denom = (float*)   carve((size_t)NN * 4);
    float*    exbuf = (float*)   carve((size_t)NE2 * 4);
    float*    scale = (float*)   carve(CC);
    float*    shiftv= (float*)   carve(CC);
    float*    sums  = (float*)   carve(NG * CC);
    float*    cnt   = (float*)   carve(NG);
    float*    gmean = (float*)   carve(NG * CC);
    float*    gq    = (float*)   carve(NG * CC);

    const dim3 blk(256);
    const unsigned mblocks = (NN + 63) / 64;           // 64-row GEMM block tiles
    const unsigned nb256   = (CC + 63) / 64;           // N=256 -> 4 column blocks
    const unsigned eh4     = ((unsigned)NE2 * 4 + 255) / 256;
    const unsigned eh1     = ((unsigned)NE2 * 1 + 255) / 256;
    const unsigned ewaves  = ((unsigned)NE2 * 32 + 255) / 256;
    const unsigned elemb   = ((unsigned)NN * CC + 255) / 256;

    // ---------- Layer 1: GATConv(128 -> 4 heads x 64) + BN + ReLU ----------
    gemm_wmma_kernel<CC, false, false, false><<<dim3(nb256, mblocks), blk, 0, stream>>>(
        x, W1, hlin, NN, 128, nullptr, nullptr, nullptr);
    alphas_kernel<<<(NN * 4 + 255) / 256, blk, 0, stream>>>(
        hlin, a_src1, a_dst1, asv, adv, NN, 4, 64);
    hipMemsetAsync(menc,  0, (size_t)NN * 4 * sizeof(unsigned), stream);
    hipMemsetAsync(denom, 0, (size_t)NN * 4 * sizeof(float),    stream);
    hipMemsetAsync(agg,   0, (size_t)NN * CC * sizeof(float),   stream);
    edge_max_kernel<<<eh4, blk, 0, stream>>>(ei, asv, adv, menc, 4);
    edge_exp_kernel<<<eh4, blk, 0, stream>>>(ei, asv, adv, menc, denom, exbuf, 4);
    edge_agg_kernel<<<ewaves, blk, 0, stream>>>(ei, hlin, exbuf, denom, agg, 4, 64);
    bn_stats_kernel<<<CC, blk, 0, stream>>>(agg, g1, beta1, scale, shiftv, NN, CC);
    bn_apply_kernel<<<elemb, blk, 0, stream>>>(agg, scale, shiftv, NN * CC, CC);
    // agg now holds h1

    // ---------- Layer 2: GATConv(256 -> 256, 1 head) + BN + ReLU ----------
    gemm_wmma_kernel<CC, false, false, false><<<dim3(nb256, mblocks), blk, 0, stream>>>(
        agg, W2, hlin, NN, CC, nullptr, nullptr, nullptr);
    alphas_kernel<<<(NN + 255) / 256, blk, 0, stream>>>(
        hlin, a_src2, a_dst2, asv, adv, NN, 1, 256);
    hipMemsetAsync(menc,  0, (size_t)NN * sizeof(unsigned), stream);
    hipMemsetAsync(denom, 0, (size_t)NN * sizeof(float),    stream);
    hipMemsetAsync(agg,   0, (size_t)NN * CC * sizeof(float), stream);   // h1 dead now
    edge_max_kernel<<<eh1, blk, 0, stream>>>(ei, asv, adv, menc, 1);
    edge_exp_kernel<<<eh1, blk, 0, stream>>>(ei, asv, adv, menc, denom, exbuf, 1);
    edge_agg_kernel<<<ewaves, blk, 0, stream>>>(ei, hlin, exbuf, denom, agg, 1, 256);
    bn_stats_kernel<<<CC, blk, 0, stream>>>(agg, g2, beta2, scale, shiftv, NN, CC);
    bn_apply_kernel<<<elemb, blk, 0, stream>>>(agg, scale, shiftv, NN * CC, CC);
    // agg now holds h2

    // ---------- Pool + Q head ----------
    hipMemsetAsync(sums, 0, NG * CC * sizeof(float), stream);
    hipMemsetAsync(cnt,  0, NG * sizeof(float),      stream);
    pool_kernel<<<mblocks, blk, 0, stream>>>(agg, batch, sums, cnt, NN);
    gmean_kernel<<<(NG * CC + 255) / 256, blk, 0, stream>>>(sums, cnt, gmean);
    // gq[g,:] = gmean[g,:] @ Wq1_bot + bq1   (folds the gmean half of ctx@Wq1)
    gemm_wmma_kernel<CC, true, false, false><<<dim3(nb256, 1), blk, 0, stream>>>(
        gmean, Wq1 + (size_t)CC * CC, gq, NG, CC, bq1, nullptr, nullptr);
    // q1 = relu(h2 @ Wq1_top + gq[batch])  -> reuse hlin
    gemm_wmma_kernel<CC, false, true, true><<<dim3(nb256, mblocks), blk, 0, stream>>>(
        agg, Wq1, hlin, NN, CC, nullptr, gq, batch);
    // q = q1 @ Wq2 + bq2  -> d_out[0 : 200000]
    gemm_wmma_kernel<4, true, false, false><<<dim3(1, mblocks), blk, 0, stream>>>(
        hlin, Wq2, qout, NN, CC, bq2, nullptr, nullptr);
    // batch_index as floats -> d_out[200000 : 250000]
    tail_kernel<<<(NN + 255) / 256, blk, 0, stream>>>(batch, qout + (size_t)NN * 4, NN);
}
